// DPSegmenter_40819369181184
// MI455X (gfx1250) — matrix operand order repaired and checked by reading, not compiled, not run on hardware
//
#include <hip/hip_runtime.h>
#include <hip/hip_bf16.h>

#define BB 16
#define TT 1024
#define DD 512
#define VV 256
#define LL 64

static constexpr float FINF = 1e30f;

typedef __attribute__((ext_vector_type(16))) _Float16 v16h;
typedef __attribute__((ext_vector_type(8)))  _Float16 v8h;
typedef __attribute__((ext_vector_type(8)))  float    v8f;

// ---------------------------------------------------------------------------
// Phase 1: f32 -> f16 conversion + per-row squared norms (rows of length 512)
// ---------------------------------------------------------------------------
__global__ __launch_bounds__(256)
void convert_norm_kernel(const float* __restrict__ src, _Float16* __restrict__ dst,
                         float* __restrict__ norms) {
  const int row = blockIdx.x;
  const int tid = threadIdx.x;
  const size_t base = (size_t)row * DD;
  float x0 = src[base + tid];
  float x1 = src[base + tid + 256];
  dst[base + tid]       = (_Float16)x0;
  dst[base + tid + 256] = (_Float16)x1;
  float ss = x0 * x0 + x1 * x1;
  #pragma unroll
  for (int off = 16; off > 0; off >>= 1) ss += __shfl_xor(ss, off, 32);
  __shared__ float part[8];
  if ((tid & 31) == 0) part[tid >> 5] = ss;
  __syncthreads();
  if (tid == 0) {
    float t = 0.f;
    #pragma unroll
    for (int w = 0; w < 8; ++w) t += part[w];
    norms[row] = t;
  }
}

// ---------------------------------------------------------------------------
// Phase 2: WMMA GEMM with async-staged codebook tile:
//   dist[m][n] = rn[m] + cn[n] - 2 * sum_k A[m][k]*Bt[n][k]
// The 16x512 B-slice (shared by all 8 waves of the block) is copied to LDS
// once via GLOBAL_LOAD_ASYNC_TO_LDS_B128 (ASYNCcnt), then read via ds_load.
// Rows padded to 1040B so consecutive rows start 4 banks apart.
// ---------------------------------------------------------------------------
#define BROW_HALFS 520   // 512 data halfs + 8 pad halfs (1040 bytes)

__global__ __launch_bounds__(256)
void gemm_dist_kernel(const _Float16* __restrict__ Ah, const _Float16* __restrict__ Bh,
                      const float* __restrict__ rn, const float* __restrict__ cn,
                      float* __restrict__ dist) {
  __shared__ _Float16 btile[16 * BROW_HALFS];

  const int tid   = threadIdx.x;
  const int lane  = tid & 31;
  const int wave  = tid >> 5;
  const int mtile = blockIdx.x * 8 + wave;   // 16-row tile of M = B*T
  const int ntile = blockIdx.y;              // 16-col tile of V (shared per block)
  const int mlo = lane & 15;
  const int hi  = lane >> 4;

  // ---- async stage: 16 rows x 512 halfs = 1024 x 16B chunks, 4 per thread ----
  {
    const unsigned ldsBase = (unsigned)(uintptr_t)(&btile[0]);
    #pragma unroll
    for (int i = 0; i < 4; ++i) {
      const int c   = tid + i * 256;   // 16B-chunk id 0..1023
      const int row = c >> 6;          // 0..15
      const int col = c & 63;          // 16B units within the row
      const void* g = (const void*)(Bh + (size_t)(ntile * 16 + row) * DD + col * 8);
      const unsigned l = ldsBase + (unsigned)(row * (BROW_HALFS * 2) + col * 16);
      asm volatile("global_load_async_to_lds_b128 %0, %1, off"
                   :: "v"(l), "v"(g) : "memory");
    }
    asm volatile("s_wait_asynccnt 0x0" ::: "memory");
  }
  __syncthreads();

  // A: lane lo group holds K=0..7 & 16..23 of row mlo; hi group K=8..15 & 24..31
  const _Float16* aptr = Ah + (size_t)(mtile * 16 + mlo) * DD + hi * 8;
  // B fragment from LDS: lo group K=0..7 & 16..23 of column n=mlo; hi group K=8..15 & 24..31
  const _Float16* bl = &btile[0] + (size_t)mlo * BROW_HALFS + hi * 8;

  v8f acc = {};
  #pragma unroll 4
  for (int k0 = 0; k0 < DD; k0 += 32) {
    v8h a0 = *(const v8h*)(aptr + k0);
    v8h a1 = *(const v8h*)(aptr + k0 + 16);
    v8h b0 = *(const v8h*)(bl + k0);
    v8h b1 = *(const v8h*)(bl + k0 + 16);
    v16h a, b;
    #pragma unroll
    for (int i = 0; i < 8; ++i) {
      a[i] = a0[i]; a[8 + i] = a1[i];
      b[i] = b0[i]; b[8 + i] = b1[i];
    }
    acc = __builtin_amdgcn_wmma_f32_16x16x32_f16(false, a, false, b,
                                                 (short)0, acc, false, false);
  }

  const int n = ntile * 16 + mlo;
  const float cnv = cn[n];
  #pragma unroll
  for (int i = 0; i < 8; ++i) {
    const int m = mtile * 16 + hi * 8 + i;   // C/D layout: VGPR i -> M = 8*hi + i
    dist[(size_t)m * VV + n] = rn[m] + cnv - 2.0f * acc[i];
  }
}

// ---------------------------------------------------------------------------
// Phase 3: cumulative cost table C[b][t][v], C[b][0][v] = 0 (L2-resident scan)
// ---------------------------------------------------------------------------
__global__ __launch_bounds__(256)
void cumsum_kernel(const float* __restrict__ dist, float* __restrict__ C) {
  const int b = blockIdx.x;
  const int v = threadIdx.x;
  const size_t dbase = (size_t)b * TT * VV + v;
  const size_t cbase = (size_t)b * (TT + 1) * VV + v;
  float acc = 0.f;
  C[cbase] = 0.f;
  for (int t = 0; t < TT; ++t) {
    if (t + 8 < TT) __builtin_prefetch(&dist[dbase + (size_t)(t + 8) * VV], 0, 0);
    acc += dist[dbase + (size_t)t * VV];
    C[cbase + (size_t)(t + 1) * VV] = acc;
  }
}

// ---------------------------------------------------------------------------
// Phase 4: per-(b,t,start) best token: segmin/segtok over V=256 candidates.
// Block = (t,b); thread = v; wave32 shfl argmin + LDS cross-wave combine.
// j indexes start s = t-1-j (s >= max(t-L,0) automatically).
// ---------------------------------------------------------------------------
__global__ __launch_bounds__(256)
void segmin_kernel(const float* __restrict__ C, float* __restrict__ segmin,
                   int* __restrict__ segtok) {
  const int t = blockIdx.x + 1;
  const int b = blockIdx.y;
  const int v = threadIdx.x;
  const float* Cb = C + (size_t)b * (TT + 1) * VV;
  const float ct = Cb[(size_t)t * VV + v];
  __shared__ float wval[8];
  __shared__ int   widx[8];
  const size_t obase = (size_t)(b * TT + (t - 1)) * LL;

  for (int j = 0; j < LL; ++j) {
    const int s = t - 1 - j;            // uniform across block
    if (s < 0) {
      if (v == 0) { segmin[obase + j] = FINF; segtok[obase + j] = 0; }
      continue;
    }
    float val = ct - Cb[(size_t)s * VV + v];
    int   idx = v;
    #pragma unroll
    for (int off = 16; off > 0; off >>= 1) {
      float ov = __shfl_xor(val, off, 32);
      int   oi = __shfl_xor(idx, off, 32);
      if (ov < val || (ov == val && oi < idx)) { val = ov; idx = oi; }
    }
    if ((v & 31) == 0) { wval[v >> 5] = val; widx[v >> 5] = idx; }
    __syncthreads();
    if (v == 0) {
      float bv = wval[0]; int bi = widx[0];
      #pragma unroll
      for (int w = 1; w < 8; ++w)
        if (wval[w] < bv || (wval[w] == bv && widx[w] < bi)) { bv = wval[w]; bi = widx[w]; }
      segmin[obase + j] = bv;
      segtok[obase + j] = bi;
    }
    __syncthreads();
  }
}

// ---------------------------------------------------------------------------
// Phase 5: serial DP over t (the only sequential part). One block per batch,
// 64 threads (one per window offset), costs held in LDS.
// ---------------------------------------------------------------------------
__global__ __launch_bounds__(64)
void dp_kernel(const float* __restrict__ segmin, const int* __restrict__ segtok,
               const int* __restrict__ lengths,
               int* __restrict__ prev, int* __restrict__ tok,
               float* __restrict__ final_costs) {
  const int b = blockIdx.x;
  const int j = threadIdx.x;
  __shared__ float costs[TT + 1];
  __shared__ float rv[2];
  __shared__ int   ri[2];
  for (int i = j; i <= TT; i += 64) costs[i] = 0.f;
  __syncthreads();

  for (int t = 1; t <= TT; ++t) {
    const size_t base = (size_t)(b * TT + (t - 1)) * LL;
    const int s = t - 1 - j;
    float total = FINF;
    if (s >= 0) total = costs[s] + segmin[base + j];
    int idx = j;
    #pragma unroll
    for (int off = 16; off > 0; off >>= 1) {
      float ov = __shfl_xor(total, off, 32);
      int   oi = __shfl_xor(idx, off, 32);
      if (ov < total || (ov == total && oi < idx)) { total = ov; idx = oi; }
    }
    if ((j & 31) == 0) { rv[j >> 5] = total; ri[j >> 5] = idx; }
    __syncthreads();
    if (j == 0) {
      float bv = rv[0]; int bi = ri[0];
      if (rv[1] < bv || (rv[1] == bv && ri[1] < bi)) { bv = rv[1]; bi = ri[1]; }
      costs[t] = bv;
      prev[b * (TT + 1) + t] = t - 1 - bi;
      tok[b * (TT + 1) + t]  = segtok[base + bi];
    }
    __syncthreads();
  }
  if (j == 0) {
    int len = lengths[b];
    if (len > TT) len = TT;
    if (len < 0)  len = 0;
    final_costs[b] = costs[len];
  }
}

// ---------------------------------------------------------------------------
// Phase 6: backtrack + reverse to front-to-back order. One lane per batch.
// ---------------------------------------------------------------------------
__global__ void backtrack_kernel(const int* __restrict__ prev, const int* __restrict__ tok,
                                 const int* __restrict__ lengths,
                                 int* __restrict__ outB, int* __restrict__ outT,
                                 int* __restrict__ outC,
                                 int* __restrict__ stB, int* __restrict__ stT) {
  const int b = threadIdx.x;
  if (b >= BB) return;
  int st = lengths[b];
  if (st > TT) st = TT;
  if (st < 0)  st = 0;
  int n = 0;
  while (st > 0 && n < TT) {
    stB[b * TT + n] = st;
    stT[b * TT + n] = tok[b * (TT + 1) + st];
    st = prev[b * (TT + 1) + st];
    ++n;
  }
  outC[b] = n;
  for (int i = 0; i < TT; ++i) {
    outB[b * TT + i] = (i < n) ? stB[b * TT + (n - 1 - i)] : 0;
    outT[b * TT + i] = (i < n) ? stT[b * TT + (n - 1 - i)] : 0;
  }
}

// ---------------------------------------------------------------------------
extern "C" void kernel_launch(void* const* d_in, const int* in_sizes, int n_in,
                              void* d_out, int out_size, void* d_ws, size_t ws_size,
                              hipStream_t stream) {
  const float* reps     = (const float*)d_in[0];   // (B,T,D) f32
  const float* codebook = (const float*)d_in[1];   // (V,D)   f32
  const int*   lengths  = (const int*)d_in[2];     // (B,)    i32

  char* ws = (char*)d_ws;
  size_t off = 0;
  auto take = [&](size_t bytes) -> void* {
    void* p = ws + off;
    off = (off + bytes + 255) & ~(size_t)255;
    return p;
  };

  _Float16* repsH  = (_Float16*)take((size_t)BB * TT * DD * sizeof(_Float16));
  _Float16* cbH    = (_Float16*)take((size_t)VV * DD * sizeof(_Float16));
  float*    rn     = (float*)take((size_t)BB * TT * sizeof(float));
  float*    cn     = (float*)take((size_t)VV * sizeof(float));
  float*    dist   = (float*)take((size_t)BB * TT * VV * sizeof(float));
  float*    Ctab   = (float*)take((size_t)BB * (TT + 1) * VV * sizeof(float));
  float*    segmin = (float*)take((size_t)BB * TT * LL * sizeof(float));
  int*      segtok = (int*)take((size_t)BB * TT * LL * sizeof(int));
  int*      prevA  = (int*)take((size_t)BB * (TT + 1) * sizeof(int));
  int*      tokA   = (int*)take((size_t)BB * (TT + 1) * sizeof(int));
  int*      stB    = (int*)take((size_t)BB * TT * sizeof(int));
  int*      stT    = (int*)take((size_t)BB * TT * sizeof(int));
  (void)ws_size; (void)n_in; (void)in_sizes; (void)out_size;

  int*   outB = (int*)d_out;
  int*   outT = outB + (size_t)BB * TT;
  int*   outC = outT + (size_t)BB * TT;
  float* outF = (float*)(outC + BB);

  // 1) f32 -> f16 + norms
  convert_norm_kernel<<<BB * TT, 256, 0, stream>>>(reps, repsH, rn);
  convert_norm_kernel<<<VV, 256, 0, stream>>>(codebook, cbH, cn);

  // 2) WMMA distance GEMM: grid (M/16/8, V/16), 8 waves per block,
  //    codebook tile async-staged to LDS
  gemm_dist_kernel<<<dim3((BB * TT) / 128, VV / 16), 256, 0, stream>>>(
      repsH, cbH, rn, cn, dist);

  // 3) cumulative costs
  cumsum_kernel<<<BB, VV, 0, stream>>>(dist, Ctab);

  // 4) per-(t,start) best token
  segmin_kernel<<<dim3(TT, BB), 256, 0, stream>>>(Ctab, segmin, segtok);

  // 5) serial DP
  dp_kernel<<<BB, 64, 0, stream>>>(segmin, segtok, lengths, prevA, tokA, outF);

  // 6) backtrack
  backtrack_kernel<<<1, 32, 0, stream>>>(prevA, tokA, lengths, outB, outT, outC,
                                         stB, stT);
}
